// TransformerBlock_56667798503734
// MI455X (gfx1250) — matrix operand order repaired and checked by reading, not compiled, not run on hardware
//
#include <hip/hip_runtime.h>
#include <hip/hip_bf16.h>
#include <math.h>

// ---------------------------------------------------------------------------
// Types for CDNA5 WMMA (wave32): D(16x16 f32) = A(16x32 f16) * B(32x16 f16) + C
// ---------------------------------------------------------------------------
typedef _Float16 f16;
typedef __attribute__((ext_vector_type(16))) _Float16 v16h;
typedef __attribute__((ext_vector_type(8)))  _Float16 v8h;
typedef __attribute__((ext_vector_type(8)))  float    v8f;

// Pointer types matching the gfx1250 async-LDS builtin signature:
//   void __builtin_amdgcn_global_load_async_to_lds_b128(int4 AS1*, int4 AS3*, imm, imm)
typedef int v4i_t __attribute__((vector_size(16)));
typedef __attribute__((address_space(1))) v4i_t* gptr_b128;
typedef __attribute__((address_space(3))) v4i_t* lptr_b128;

#if defined(__HIP_DEVICE_COMPILE__) && __has_builtin(__builtin_amdgcn_global_load_async_to_lds_b128)
#define ATHENA_ASYNC_LDS 1
#else
#define ATHENA_ASYNC_LDS 0
#endif

__device__ __forceinline__ v16h cat8(v8h lo, v8h hi) {
    return __builtin_shufflevector(lo, hi, 0,1,2,3,4,5,6,7,8,9,10,11,12,13,14,15);
}

__device__ __forceinline__ v8f wmma_f16(v16h a, v16h b, v8f c) {
    // v_wmma_f32_16x16x32_f16  (neg_a, A, neg_b, B, c_mod, C, reuse_a, reuse_b)
    return __builtin_amdgcn_wmma_f32_16x16x32_f16(false, a, false, b, (short)0, c, false, false);
}

// 16-byte global -> LDS copy.  Async (ASYNCcnt, no VGPR data path) when the
// gfx1250 builtin is available; plain load/store otherwise.
__device__ __forceinline__ void cp16_g2l(const f16* g, f16* l) {
#if ATHENA_ASYNC_LDS
    __builtin_amdgcn_global_load_async_to_lds_b128(
        (gptr_b128)(unsigned long long)g,
        (lptr_b128)(unsigned int)(unsigned long long)l,   // low 32 bits = LDS offset
        0, 0);
#else
    *(v8h*)l = *(const v8h*)g;
#endif
}

// Wait for all but the newest async tile (4 copies/thread) to land.
__device__ __forceinline__ void wait_g2l(bool keep4) {
#if ATHENA_ASYNC_LDS
  #if __has_builtin(__builtin_amdgcn_s_wait_asynccnt)
    if (keep4) __builtin_amdgcn_s_wait_asynccnt(4);
    else       __builtin_amdgcn_s_wait_asynccnt(0);
  #else
    if (keep4) __asm__ volatile("s_wait_asynccnt 4" ::: "memory");
    else       __asm__ volatile("s_wait_asynccnt 0" ::: "memory");
  #endif
#endif
}

// A-fragment (16x32, row M = lane&15): halves 0-7 = K kb..kb+7, 8-15 = kb+16..kb+23,
// kb = 0 (lanes 0-15) or 8 (lanes 16-31).  ld in halves.
__device__ __forceinline__ v16h load_a_frag(const f16* base, int ld) {
    const int lane = threadIdx.x & 31;
    const int row  = lane & 15;
    const int kb   = (lane < 16) ? 0 : 8;
    const f16* p = base + row * ld + kb;
    return cat8(*(const v8h*)(p), *(const v8h*)(p + 16));
}

// B-fragment (32x16, col N = lane&15): halves 0-7 = K kb..kb+7, 8-15 = kb+16..kb+23, kb = 0 / 8.
// Source is Bt: [N][K] row-major.  ld in halves.
__device__ __forceinline__ v16h load_b_frag(const f16* base, int ld) {
    const int lane = threadIdx.x & 31;
    const int col  = lane & 15;
    const int kb   = (lane < 16) ? 0 : 8;
    const f16* p = base + col * ld + kb;
    return cat8(*(const v8h*)(p), *(const v8h*)(p + 16));
}

// Half-wave (16-lane) reductions: C/D rows live in lanes 0-15 (M 0-7) / 16-31 (M 8-15).
__device__ __forceinline__ float redmax16(float v) {
    #pragma unroll
    for (int d = 1; d < 16; d <<= 1) v = fmaxf(v, __shfl_xor(v, d, 32));
    return v;
}
__device__ __forceinline__ float redsum16(float v) {
    #pragma unroll
    for (int d = 1; d < 16; d <<= 1) v += __shfl_xor(v, d, 32);
    return v;
}

// ---------------------------------------------------------------------------
// Weight transpose + f16 convert:  W[K,N] f32  ->  Wt[N,K] f16
// ---------------------------------------------------------------------------
__global__ __launch_bounds__(256) void wt_to_f16t(const float* __restrict__ W,
                                                  f16* __restrict__ Wt,
                                                  int Kd, int Nd) {
    int i = blockIdx.x * 256 + threadIdx.x;
    if (i < Kd * Nd) {
        int k = i / Nd, n = i % Nd;
        Wt[(size_t)n * Kd + k] = (f16)W[i];
    }
}

// ---------------------------------------------------------------------------
// LayerNorm over 512 cols -> f16.  One 256-thread block per row.
// ---------------------------------------------------------------------------
__global__ __launch_bounds__(256) void layernorm_f16(const float* __restrict__ x,
                                                     const float* __restrict__ g,
                                                     const float* __restrict__ be,
                                                     f16* __restrict__ out) {
    __shared__ float red[256];
    const int tid = threadIdx.x;
    const size_t row = blockIdx.x;
    const float* xr = x + row * 512;
    float v0 = xr[tid], v1 = xr[tid + 256];

    red[tid] = v0 + v1;
    __syncthreads();
    #pragma unroll
    for (int s = 128; s > 0; s >>= 1) { if (tid < s) red[tid] += red[tid + s]; __syncthreads(); }
    float mu = red[0] * (1.0f / 512.0f);
    __syncthreads();

    float d0 = v0 - mu, d1 = v1 - mu;
    red[tid] = d0 * d0 + d1 * d1;
    __syncthreads();
    #pragma unroll
    for (int s = 128; s > 0; s >>= 1) { if (tid < s) red[tid] += red[tid + s]; __syncthreads(); }
    float rs = rsqrtf(red[0] * (1.0f / 512.0f) + 1e-5f);

    out[row * 512 + tid]       = (f16)(d0 * rs * g[tid]       + be[tid]);
    out[row * 512 + tid + 256] = (f16)(d1 * rs * g[tid + 256] + be[tid + 256]);
}

// ---------------------------------------------------------------------------
// WMMA GEMM:  C[M,N] = A[M,K](f16) * Bt[N,K]^T(f16) + bias (+res) (gelu?)
// Block: 256 threads (8 waves 4x2), tile 128x128, K-step 32.
// Wave tile 32x64 -> 8 WMMAs / wave / step.  Double-buffered async LDS staging.
// ---------------------------------------------------------------------------
template <bool RES, bool GELU, bool F16OUT>
__global__ __launch_bounds__(256) void gemm_wmma(const f16* __restrict__ A,
                                                 const f16* __restrict__ Bt,
                                                 const float* __restrict__ bias,
                                                 const float* __restrict__ res,
                                                 float* __restrict__ Cout,
                                                 f16* __restrict__ C16,
                                                 int M, int N, int K) {
    __shared__ f16 As[2][128 * 40];
    __shared__ f16 Bs[2][128 * 40];
    const int tid  = threadIdx.x;
    const int lane = tid & 31;
    const int w    = tid >> 5;
    const int wr   = w >> 1;   // 0..3 -> 32-row strip
    const int wc   = w & 1;    // 0..1 -> 64-col strip
    const int m0   = blockIdx.y * 128;
    const int n0   = blockIdx.x * 128;

    // Per-thread staging coords: 512 16B-chunks per 128x32 tile, 2 per thread.
    const int r0 = (0 * 256 + tid) >> 2, kc0 = ((0 * 256 + tid) & 3) * 8;
    const int r1 = (1 * 256 + tid) >> 2, kc1 = ((1 * 256 + tid) & 3) * 8;

    auto stage = [&](int buf, int k0) {
        cp16_g2l(&A [(size_t)(m0 + r0) * K + k0 + kc0], &As[buf][r0 * 40 + kc0]);
        cp16_g2l(&A [(size_t)(m0 + r1) * K + k0 + kc1], &As[buf][r1 * 40 + kc1]);
        cp16_g2l(&Bt[(size_t)(n0 + r0) * K + k0 + kc0], &Bs[buf][r0 * 40 + kc0]);
        cp16_g2l(&Bt[(size_t)(n0 + r1) * K + k0 + kc1], &Bs[buf][r1 * 40 + kc1]);
    };

    v8f acc[2][4] = {};
    const int nt = K >> 5;

    stage(0, 0);
    for (int t = 0; t < nt; ++t) {
        if (t + 1 < nt) { stage((t + 1) & 1, (t + 1) * 32); wait_g2l(true); }
        else            { wait_g2l(false); }
        __syncthreads();

        const f16* as = &As[t & 1][0];
        const f16* bs = &Bs[t & 1][0];
        v16h a[2], b[4];
        #pragma unroll
        for (int i = 0; i < 2; ++i) a[i] = load_a_frag(&as[(wr * 32 + i * 16) * 40], 40);
        #pragma unroll
        for (int j = 0; j < 4; ++j) b[j] = load_b_frag(&bs[(wc * 64 + j * 16) * 40], 40);
        #pragma unroll
        for (int i = 0; i < 2; ++i)
            #pragma unroll
            for (int j = 0; j < 4; ++j)
                acc[i][j] = wmma_f16(a[i], b[j], acc[i][j]);
        __syncthreads();
    }

    // Epilogue: lanes 0-15 hold rows 0-7 (col=lane), lanes 16-31 rows 8-15.
    const int colL  = lane & 15;
    const int rbase = (lane < 16) ? 0 : 8;
    #pragma unroll
    for (int i = 0; i < 2; ++i) {
        #pragma unroll
        for (int j = 0; j < 4; ++j) {
            int gn = n0 + wc * 64 + j * 16 + colL;
            float bv = bias[gn];
            #pragma unroll
            for (int r = 0; r < 8; ++r) {
                int gm = m0 + wr * 32 + i * 16 + rbase + r;
                float v = acc[i][j][r] + bv;
                if (GELU) v = 0.5f * v * (1.0f + erff(v * 0.70710678118654752f));
                if (RES)  v += res[(size_t)gm * N + gn];
                if (F16OUT) C16[(size_t)gm * N + gn] = (f16)v;
                else        Cout[(size_t)gm * N + gn] = v;
            }
        }
    }
}

// ---------------------------------------------------------------------------
// Repack QKV f16 [8192,512] -> per-head layouts.
// mode 0: dst[b,h,l,d] (Q scaled 1/8, K scale 1)   mode 1: dst[b,h,d,l] (V^T)
// ---------------------------------------------------------------------------
__global__ __launch_bounds__(256) void repack_qkv(const f16* __restrict__ src,
                                                  f16* __restrict__ dst,
                                                  float scale, int mode) {
    int i = blockIdx.x * 256 + threadIdx.x;
    if (i >= 8192 * 512) return;
    int tok = i >> 9, c = i & 511;
    int b = tok >> 11, l = tok & 2047;
    int h = c >> 6, d = c & 63;
    f16 v = (f16)((float)src[i] * scale);
    if (mode == 0) dst[((size_t)(b * 8 + h) * 2048 + l) * 64 + d] = v;
    else           dst[((size_t)(b * 8 + h) * 64 + d) * 2048 + l] = v;
}

// ---------------------------------------------------------------------------
// Flash attention (causal + pad mask).  Grid: (L/64, B*H), block 128 (4 waves).
// Wave w handles 16 queries; streams keys in 32-chunks; online softmax.
// Qh,Kh: [B,H,L,64] f16 (Q pre-scaled 1/8).  Vt: [B,H,64,L] f16.
// Out: Oc16 [B*L, 512] f16  (col = h*64+d).
// ---------------------------------------------------------------------------
__global__ __launch_bounds__(128) void flash_attn(const f16* __restrict__ Qh,
                                                  const f16* __restrict__ Kh,
                                                  const f16* __restrict__ Vt,
                                                  const int* __restrict__ mask,
                                                  f16* __restrict__ Oc16) {
    __shared__ f16 Ps[4][16 * 40];   // per-wave P staging (16 q x 32 k)
    const int tid  = threadIdx.x;
    const int lane = tid & 31;
    const int w    = tid >> 5;
    const int bh   = blockIdx.y;           // 0..31
    const int b    = bh >> 3, h = bh & 7;
    const int L = 2048, DK = 64;
    const int qbase = blockIdx.x * 64 + w * 16;

    const f16* Qb = Qh + (size_t)bh * L * DK;
    const f16* Kb = Kh + (size_t)bh * L * DK;
    const f16* Vb = Vt + (size_t)bh * DK * L;
    const int* mb = mask + b * L;

    const int colL  = lane & 15;
    const int rbase = (lane < 16) ? 0 : 8;
    const int akb   = (lane < 16) ? 0 : 8;    // A-frag K offset (gap pattern)
    const int bkb   = (lane < 16) ? 0 : 8;    // B-frag K offset (gap pattern)

    // Q A-fragments over d = [0,32) and [32,64)
    v16h aQ[2];
    {
        const f16* qp = Qb + (size_t)(qbase + (lane & 15)) * DK;
        #pragma unroll
        for (int c = 0; c < 2; ++c) {
            const f16* p = qp + c * 32 + akb;
            aQ[c] = cat8(*(const v8h*)p, *(const v8h*)(p + 16));
        }
    }

    v8f accO[4] = {};
    float mrow[8], lrow[8];
    #pragma unroll
    for (int r = 0; r < 8; ++r) { mrow[r] = -3.0e38f; lrow[r] = 0.0f; }

    for (int ks = 0; ks < qbase + 16; ks += 32) {
        // ---- scores: S[16q x 32k] = Q (16x64) * K^T, 4 WMMAs ----
        v8f sc[2] = {};
        #pragma unroll
        for (int j = 0; j < 2; ++j) {
            #pragma unroll
            for (int c = 0; c < 2; ++c) {
                const f16* p = Kb + (size_t)(ks + j * 16 + colL) * DK + c * 32 + bkb;
                v16h bk = cat8(*(const v8h*)p, *(const v8h*)(p + 16));
                sc[j] = wmma_f16(aQ[c], bk, sc[j]);
            }
        }

        const int kid0 = ks + colL, kid1 = ks + 16 + colL;
        const int mv0 = mb[kid0], mv1 = mb[kid1];

        // ---- online softmax + stage P(f16) into LDS ----
        #pragma unroll
        for (int r = 0; r < 8; ++r) {
            const int qrow = qbase + rbase + r;
            float s0 = ((kid0 <= qrow) && (mv0 != 0)) ? sc[0][r] : -1.0e9f;
            float s1 = ((kid1 <= qrow) && (mv1 != 0)) ? sc[1][r] : -1.0e9f;
            float mx = redmax16(fmaxf(s0, s1));
            float mnew = fmaxf(mrow[r], mx);
            float alpha = __expf(mrow[r] - mnew);
            float p0 = __expf(s0 - mnew);
            float p1 = __expf(s1 - mnew);
            float psum = redsum16(p0 + p1);
            lrow[r] = lrow[r] * alpha + psum;
            mrow[r] = mnew;
            #pragma unroll
            for (int jd = 0; jd < 4; ++jd) accO[jd][r] *= alpha;
            Ps[w][(rbase + r) * 40 + colL]      = (f16)p0;
            Ps[w][(rbase + r) * 40 + 16 + colL] = (f16)p1;
        }
        __asm__ volatile("s_wait_dscnt 0" ::: "memory");

        // ---- O += P (16x32) * V (32x64), 4 WMMAs ----
        v16h aP = load_a_frag(&Ps[w][0], 40);
        #pragma unroll
        for (int jd = 0; jd < 4; ++jd) {
            const f16* p = Vb + (size_t)(jd * 16 + colL) * L + ks + bkb;
            v16h bv = cat8(*(const v8h*)p, *(const v8h*)(p + 16));
            accO[jd] = wmma_f16(aP, bv, accO[jd]);
        }
    }

    // ---- write O / l  (f16, [B*L, 512]) ----
    #pragma unroll
    for (int jd = 0; jd < 4; ++jd) {
        #pragma unroll
        for (int r = 0; r < 8; ++r) {
            int q = qbase + rbase + r;
            float ov = (lrow[r] > 0.0f) ? (accO[jd][r] / lrow[r]) : 0.0f;
            Oc16[(size_t)(b * L + q) * 512 + h * 64 + jd * 16 + colL] = (f16)ov;
        }
    }
}

// ---------------------------------------------------------------------------
// Host-side orchestration
// ---------------------------------------------------------------------------
extern "C" void kernel_launch(void* const* d_in, const int* in_sizes, int n_in,
                              void* d_out, int out_size, void* d_ws, size_t ws_size,
                              hipStream_t stream) {
    const float* x    = (const float*)d_in[0];
    const int*   mask = (const int*)  d_in[1];
    const float* Wq = (const float*)d_in[2];  const float* bq = (const float*)d_in[3];
    const float* Wk = (const float*)d_in[4];  const float* bk = (const float*)d_in[5];
    const float* Wv = (const float*)d_in[6];  const float* bv = (const float*)d_in[7];
    const float* Wo = (const float*)d_in[8];  const float* bo = (const float*)d_in[9];
    const float* g1 = (const float*)d_in[10]; const float* be1 = (const float*)d_in[11];
    const float* g2 = (const float*)d_in[12]; const float* be2 = (const float*)d_in[13];
    const float* W1 = (const float*)d_in[14]; const float* b1 = (const float*)d_in[15];
    const float* W2 = (const float*)d_in[16]; const float* b2 = (const float*)d_in[17];

    const int M = 8192, D = 512, DFF = 2048;

    // Carve workspace
    char* base = (char*)d_ws; size_t off = 0;
    auto take = [&](size_t bytes) -> char* {
        char* p = base + off; off += (bytes + 255) & ~(size_t)255; return p;
    };
    f16* Wqt = (f16*)take((size_t)D * D * 2);
    f16* Wkt = (f16*)take((size_t)D * D * 2);
    f16* Wvt = (f16*)take((size_t)D * D * 2);
    f16* Wot = (f16*)take((size_t)D * D * 2);
    f16* W1t = (f16*)take((size_t)DFF * D * 2);   // [2048 x 512]
    f16* W2t = (f16*)take((size_t)D * DFF * 2);   // [512 x 2048]
    f16* h16 = (f16*)take((size_t)M * D * 2);     // LN1 out; reused for LN2 out
    f16* q16 = (f16*)take((size_t)M * D * 2);     // reused as Oc16 after repack
    f16* k16 = (f16*)take((size_t)M * D * 2);
    f16* v16 = (f16*)take((size_t)M * D * 2);
    f16* Qh  = (f16*)take((size_t)M * D * 2);
    f16* Khd = (f16*)take((size_t)M * D * 2);
    f16* Vt  = (f16*)take((size_t)M * D * 2);
    float* x2   = (float*)take((size_t)M * D * 4);
    f16* act16  = (f16*)take((size_t)M * DFF * 2);
    f16* Oc16   = q16;
    f16* h2_16  = h16;

    // 1) Weights -> transposed f16
    wt_to_f16t<<<(D * D + 255) / 256, 256, 0, stream>>>(Wq, Wqt, D, D);
    wt_to_f16t<<<(D * D + 255) / 256, 256, 0, stream>>>(Wk, Wkt, D, D);
    wt_to_f16t<<<(D * D + 255) / 256, 256, 0, stream>>>(Wv, Wvt, D, D);
    wt_to_f16t<<<(D * D + 255) / 256, 256, 0, stream>>>(Wo, Wot, D, D);
    wt_to_f16t<<<(D * DFF + 255) / 256, 256, 0, stream>>>(W1, W1t, D, DFF);
    wt_to_f16t<<<(D * DFF + 255) / 256, 256, 0, stream>>>(W2, W2t, DFF, D);

    // 2) LN1
    layernorm_f16<<<M, 256, 0, stream>>>(x, g1, be1, h16);

    // 3) QKV projections (f16 out)
    dim3 gD(D / 128, M / 128);       // (4, 64)
    gemm_wmma<false, false, true><<<gD, 256, 0, stream>>>(h16, Wqt, bq, nullptr, nullptr, q16, M, D, D);
    gemm_wmma<false, false, true><<<gD, 256, 0, stream>>>(h16, Wkt, bk, nullptr, nullptr, k16, M, D, D);
    gemm_wmma<false, false, true><<<gD, 256, 0, stream>>>(h16, Wvt, bv, nullptr, nullptr, v16, M, D, D);

    // 4) Repack per-head (Q scaled by 1/sqrt(64)=0.125)
    const int NE = M * D;
    repack_qkv<<<(NE + 255) / 256, 256, 0, stream>>>(q16, Qh, 0.125f, 0);
    repack_qkv<<<(NE + 255) / 256, 256, 0, stream>>>(k16, Khd, 1.0f, 0);
    repack_qkv<<<(NE + 255) / 256, 256, 0, stream>>>(v16, Vt, 1.0f, 1);

    // 5) Flash attention -> Oc16 (aliases q16, safe: q16 consumed by repack above)
    flash_attn<<<dim3(2048 / 64, 32), 128, 0, stream>>>(Qh, Khd, Vt, mask, Oc16);

    // 6) O projection + residual(x) -> x2
    gemm_wmma<true, false, false><<<gD, 256, 0, stream>>>(Oc16, Wot, bo, x, x2, nullptr, M, D, D);

    // 7) LN2
    layernorm_f16<<<M, 256, 0, stream>>>(x2, g2, be2, h2_16);

    // 8) FFN1 + exact GELU (f16 out)
    dim3 gF(DFF / 128, M / 128);     // (16, 64)
    gemm_wmma<false, true, true><<<gF, 256, 0, stream>>>(h2_16, W1t, b1, nullptr, nullptr, act16, M, DFF, D);

    // 9) FFN2 + residual(x2) -> d_out
    gemm_wmma<true, false, false><<<gD, 256, 0, stream>>>(act16, W2t, b2, x2, (float*)d_out, nullptr, M, D, DFF);
}
